// _MultiheadAttention_56358560858447
// MI455X (gfx1250) — compile-verified, matrix operand order checked by reading, not run on hardware
//
#include <hip/hip_runtime.h>

typedef __attribute__((ext_vector_type(16))) __bf16 v16bf;
typedef __attribute__((ext_vector_type(8)))  float  v8f;
typedef __attribute__((ext_vector_type(4)))  float  f32x4;

constexpr int B_  = 2;
constexpr int S_  = 4096;
constexpr int D_  = 512;
constexpr int H_  = 8;
constexpr int DH_ = 64;
constexpr int M_  = B_ * S_;          // 8192 rows for the projections
constexpr float SCALE_ = 0.125f;      // DH^-0.5

// K index inside a 16x32 bf16 A/B fragment for element e (0..15) of lane-half `half`.
// ISA 7.12.2: VGPR j<4 -> K = 2j+h (+8 for lanes 16-31); j>=4 -> K = 16+2(j-4)+h (+8).
__device__ __forceinline__ int kmap(int e, int half) {
    int j = e >> 1, h = e & 1;
    int k = (j < 4) ? (2 * j + h) : (16 + 2 * (j - 4) + h);
    return k + 8 * half;
}

__device__ __forceinline__ v16bf make_frag(const float* row, int half) {
    v16bf f;
#pragma unroll
    for (int e = 0; e < 16; ++e) f[e] = (__bf16)row[kmap(e, half)];
    return f;
}
__device__ __forceinline__ v16bf make_frag(const __bf16* row, int half) {
    v16bf f;
#pragma unroll
    for (int e = 0; e < 16; ++e) f[e] = row[kmap(e, half)];
    return f;
}
// B fragment from K-major storage: element (k) at col[k*ld]
__device__ __forceinline__ v16bf make_frag_strided(const __bf16* col, int ld, int half) {
    v16bf f;
#pragma unroll
    for (int e = 0; e < 16; ++e) f[e] = col[(size_t)kmap(e, half) * ld];
    return f;
}

__device__ __forceinline__ v8f wmma_bf16(v16bf a, v16bf b, v8f c) {
    return __builtin_amdgcn_wmma_f32_16x16x32_bf16(false, a, false, b, (short)0, c,
                                                   false, false);
}

// ---------------------------------------------------------------------------
// Y[m,n] = sum_k X[m,k] * W[n,k] + bias[n].   Wave computes one 16x16 tile.
// Block = 128 threads (4 waves) covering 16(M) x 64(N).
// ---------------------------------------------------------------------------
template <typename TX, bool BF16OUT>
__global__ void gemm_xwT(const TX* __restrict__ X, const float* __restrict__ W,
                         const float* __restrict__ bias, void* __restrict__ Yv,
                         int M, int N, int K) {
    int lane = threadIdx.x & 31;
    int wid  = threadIdx.x >> 5;
    int half = lane >> 4;
    int lidx = lane & 15;
    int m0 = blockIdx.x * 16;
    int n0 = blockIdx.y * 64 + wid * 16;

    const TX*    xrow = X + (size_t)(m0 + lidx) * K;
    const float* wrow = W + (size_t)(n0 + lidx) * K;

    v8f acc = {};
    for (int k = 0; k < K; k += 32) {
        v16bf a = make_frag(xrow + k, half);
        v16bf b = make_frag(wrow + k, half);
        acc = wmma_bf16(a, b, acc);
    }
    float bb = bias[n0 + lidx];
    if constexpr (BF16OUT) {
        __bf16* Y = (__bf16*)Yv;
#pragma unroll
        for (int r = 0; r < 8; ++r)
            Y[(size_t)(m0 + r + 8 * half) * N + n0 + lidx] = (__bf16)(acc[r] + bb);
    } else {
        float* Y = (float*)Yv;
#pragma unroll
        for (int r = 0; r < 8; ++r)
            Y[(size_t)(m0 + r + 8 * half) * N + n0 + lidx] = acc[r] + bb;
    }
}

// ---------------------------------------------------------------------------
// Pass 1: per (b,h, sq-tile of 16) wave: rowmax + sumexp of scaled scores.
// Two sweeps (max-only, then sumexp) -> no serial online-softmax chain.
// ---------------------------------------------------------------------------
__global__ void attn_pass1(const __bf16* __restrict__ qb, const __bf16* __restrict__ kb,
                           float* __restrict__ rowmax, float* __restrict__ rowsum) {
    int lane = threadIdx.x & 31;
    int wid  = threadIdx.x >> 5;
    int wg   = blockIdx.x * (blockDim.x >> 5) + wid;   // 0 .. B*H*S/16-1
    int sqt  = wg % (S_ / 16);
    int bh   = wg / (S_ / 16);
    int h = bh % H_, b = bh / H_;
    int half = lane >> 4, lidx = lane & 15;
    int sq0 = sqt * 16;

    const __bf16* qrow = qb + ((size_t)(b * S_ + sq0 + lidx)) * D_ + h * DH_;
    const __bf16* kbase = kb + ((size_t)(b * S_ + lidx)) * D_ + h * DH_;
    v16bf aq0 = make_frag(qrow, half);
    v16bf aq1 = make_frag(qrow + 32, half);

    // ---- sweep 1: raw max of un-scaled scores (scale > 0 commutes with max)
    float mr[8];
#pragma unroll
    for (int r = 0; r < 8; ++r) mr[r] = -1e30f;

    for (int skt = 0; skt < S_ / 16; ++skt) {
        const __bf16* krow = kbase + (size_t)(skt * 16) * D_;
        v8f c = {};
        c = wmma_bf16(aq0, make_frag(krow, half), c);
        c = wmma_bf16(aq1, make_frag(krow + 32, half), c);
#pragma unroll
        for (int r = 0; r < 8; ++r) mr[r] = fmaxf(mr[r], c[r]);
    }
#pragma unroll
    for (int off = 1; off < 16; off <<= 1)
#pragma unroll
        for (int r = 0; r < 8; ++r)
            mr[r] = fmaxf(mr[r], __shfl_xor(mr[r], off, 32));
#pragma unroll
    for (int r = 0; r < 8; ++r) mr[r] *= SCALE_;   // scaled row max, same in all 16 lanes

    // ---- sweep 2: sumexp against known row max
    float lr[8];
#pragma unroll
    for (int r = 0; r < 8; ++r) lr[r] = 0.f;

    for (int skt = 0; skt < S_ / 16; ++skt) {
        const __bf16* krow = kbase + (size_t)(skt * 16) * D_;
        v8f c = {};
        c = wmma_bf16(aq0, make_frag(krow, half), c);
        c = wmma_bf16(aq1, make_frag(krow + 32, half), c);
#pragma unroll
        for (int r = 0; r < 8; ++r) lr[r] += __expf(fmaf(c[r], SCALE_, -mr[r]));
    }
#pragma unroll
    for (int off = 1; off < 16; off <<= 1)
#pragma unroll
        for (int r = 0; r < 8; ++r) lr[r] += __shfl_xor(lr[r], off, 32);

    if (lidx == 0) {
#pragma unroll
        for (int r = 0; r < 8; ++r) {
            int row = sq0 + r + 8 * half;
            size_t i = (size_t)(b * H_ + h) * S_ + row;
            rowmax[i] = mr[r];
            rowsum[i] = lr[r];
        }
    }
}

// ---------------------------------------------------------------------------
// Pass 2: recompute scores, normalize -> write attn (f32, nontemporal, 128B
// coalesced via ping-pong LDS restage), and ctx = attn @ V.
// Block = 128 threads (4 waves), one wave per (b,h,sq-tile).
// ---------------------------------------------------------------------------
__global__ void attn_pass2(const __bf16* __restrict__ qb, const __bf16* __restrict__ kb,
                           const __bf16* __restrict__ vb,
                           const float* __restrict__ rowmax, const float* __restrict__ rowsum,
                           float* __restrict__ attn, __bf16* __restrict__ ctx) {
    __shared__ __bf16 pbufb[2][4][16][32];   // bf16 copy -> A fragments for p@V
    __shared__ float  pbuf32[2][4][16][32];  // f32 copy  -> coalesced attn stores

    int lane = threadIdx.x & 31;
    int wid  = threadIdx.x >> 5;
    int wg   = blockIdx.x * (blockDim.x >> 5) + wid;
    int sqt  = wg % (S_ / 16);
    int bh   = wg / (S_ / 16);
    int h = bh % H_, b = bh / H_;
    int half = lane >> 4, lidx = lane & 15;
    int sq0 = sqt * 16;

    const __bf16* qrow = qb + ((size_t)(b * S_ + sq0 + lidx)) * D_ + h * DH_;
    v16bf aq0 = make_frag(qrow, half);
    v16bf aq1 = make_frag(qrow + 32, half);

    float mr[8], il[8];
#pragma unroll
    for (int r = 0; r < 8; ++r) {
        int row = sq0 + r + 8 * half;
        size_t i = (size_t)(b * H_ + h) * S_ + row;
        mr[r] = rowmax[i];
        il[r] = 1.f / rowsum[i];
    }

    // attn store mapping: 4 row-groups of 4 rows; lanes 0-7 cover one full
    // 128B row segment (32 consecutive f32) per group.
    int srow = lane >> 3;        // 0..3 row within group
    int scol = (lane & 7) * 4;   // 0,4,..,28
    float* attnRowBase =
        attn + ((size_t)(b * H_ + h) * S_ + sq0) * S_;   // + row*S_ + chunk + scol

    v8f acc[4];
#pragma unroll
    for (int t = 0; t < 4; ++t) acc[t] = (v8f){};

    for (int chunk = 0; chunk < S_; chunk += 32) {
        int pp = (chunk >> 5) & 1;   // ping-pong: WAR vs in-flight reads is safe
#pragma unroll
        for (int sub = 0; sub < 2; ++sub) {
            int sk0 = chunk + sub * 16;
            const __bf16* krow = kb + ((size_t)(b * S_ + sk0 + lidx)) * D_ + h * DH_;
            v8f c = {};
            c = wmma_bf16(aq0, make_frag(krow, half), c);
            c = wmma_bf16(aq1, make_frag(krow + 32, half), c);
#pragma unroll
            for (int r = 0; r < 8; ++r) {
                float p = __expf(fmaf(c[r], SCALE_, -mr[r])) * il[r];
                int rt = r + 8 * half;
                pbuf32[pp][wid][rt][sub * 16 + lidx] = p;
                pbufb[pp][wid][rt][sub * 16 + lidx] = (__bf16)p;
            }
        }
        asm volatile("s_wait_dscnt 0x0" ::: "memory");   // RAW: stores -> reads

        // 4 x 128-bit nontemporal stores: full 128B cachelines of attn.
#pragma unroll
        for (int g = 0; g < 4; ++g) {
            int rt = g * 4 + srow;
            f32x4 v4 = *(const f32x4*)&pbuf32[pp][wid][rt][scol];
            __builtin_nontemporal_store(
                v4, (f32x4*)(attnRowBase + (size_t)rt * S_ + chunk + scol));
        }

        // A fragment (16x32) of probabilities for p @ V
        v16bf ap;
#pragma unroll
        for (int e = 0; e < 16; ++e) ap[e] = pbufb[pp][wid][lidx][kmap(e, half)];

        const __bf16* vcol = vb + ((size_t)(b * S_ + chunk)) * D_ + h * DH_ + lidx;
#pragma unroll
        for (int t = 0; t < 4; ++t) {
            v16bf bvf = make_frag_strided(vcol + t * 16, D_, half);
            acc[t] = wmma_bf16(ap, bvf, acc[t]);
        }
    }
#pragma unroll
    for (int t = 0; t < 4; ++t)
#pragma unroll
        for (int r = 0; r < 8; ++r) {
            int row = sq0 + r + 8 * half;
            ctx[((size_t)(b * S_ + row)) * D_ + h * DH_ + t * 16 + lidx] = (__bf16)acc[t][r];
        }
}

// ---------------------------------------------------------------------------
extern "C" void kernel_launch(void* const* d_in, const int* in_sizes, int n_in,
                              void* d_out, int out_size, void* d_ws, size_t ws_size,
                              hipStream_t stream) {
    const float* Q  = (const float*)d_in[0];
    const float* Wq = (const float*)d_in[1];
    const float* bq = (const float*)d_in[2];
    const float* Wk = (const float*)d_in[3];
    const float* bk = (const float*)d_in[4];
    const float* Wv = (const float*)d_in[5];
    const float* bv = (const float*)d_in[6];
    const float* Wo = (const float*)d_in[7];
    const float* bo = (const float*)d_in[8];

    float* out  = (float*)d_out;
    float* attn = out + (size_t)B_ * S_ * D_;

    char* ws = (char*)d_ws;
    __bf16* qb  = (__bf16*)ws; ws += (size_t)M_ * D_ * sizeof(__bf16);
    __bf16* kb  = (__bf16*)ws; ws += (size_t)M_ * D_ * sizeof(__bf16);
    __bf16* vbp = (__bf16*)ws; ws += (size_t)M_ * D_ * sizeof(__bf16);
    __bf16* ctx = (__bf16*)ws; ws += (size_t)M_ * D_ * sizeof(__bf16);
    float* rmax = (float*)ws;  ws += (size_t)B_ * H_ * S_ * sizeof(float);
    float* rsum = (float*)ws;  ws += (size_t)B_ * H_ * S_ * sizeof(float);

    dim3 ggrid(M_ / 16, D_ / 64);
    gemm_xwT<float, true><<<ggrid, 128, 0, stream>>>(Q, Wq, bq, qb,  M_, D_, D_);
    gemm_xwT<float, true><<<ggrid, 128, 0, stream>>>(Q, Wk, bk, kb,  M_, D_, D_);
    gemm_xwT<float, true><<<ggrid, 128, 0, stream>>>(Q, Wv, bv, vbp, M_, D_, D_);

    int nwaves = B_ * H_ * (S_ / 16);                   // 4096
    attn_pass1<<<nwaves / 8, 256, 0, stream>>>(qb, kb, rmax, rsum);
    attn_pass2<<<nwaves / 4, 128, 0, stream>>>(qb, kb, vbp, rmax, rsum, attn, ctx);

    gemm_xwT<__bf16, false><<<ggrid, 128, 0, stream>>>(ctx, Wo, bo, out, M_, D_, D_);
}
